// FourConvBlock_14242111553633
// MI455X (gfx1250) — matrix operand order repaired and checked by reading, not compiled, not run on hardware
//
#include <hip/hip_runtime.h>
#include <math.h>

// MI455X / gfx1250, wave32. FeaStConv x4 + BatchNorm.
// Per-node WMMA GEMM computes fused [y | a] = x @ [W | u] into a padded row
// (stride S = ceil((H*Fo+H)/16)*16). B operand is pre-packed per layer into
// WMMA B-lane layout so the GEMM does only coalesced b128 loads + WMMA.
// Edge phase: L2-resident gather/softmax/scatter; self-loops folded into
// finalize (q = softmax(c) is a constant there).

typedef __attribute__((ext_vector_type(16))) _Float16 v16h;
typedef __attribute__((ext_vector_type(8)))  float    v8f;

#define HEADS 4

__global__ __launch_bounds__(256)
void fill_f32(float* __restrict__ p, int n, float v) {
  int i = blockIdx.x * blockDim.x + threadIdx.x;
  if (i < n) p[i] = v;
}

// in-degree over dst + 1.0 for the self-loop (cnt pre-filled with 1.0f)
__global__ __launch_bounds__(256)
void count_deg(const int* __restrict__ ei, int E, float* __restrict__ cnt) {
  int e = blockIdx.x * blockDim.x + threadIdx.x;
  if (e < E) atomicAdd(&cnt[ei[E + e]], 1.0f);
}

// Pack [W | u] (K padded to 32, cols padded to CT*16) into WMMA B-lane layout:
// pbuf[(ct*32 + lane)*8 + v] = {f16 B(k,col), f16 B(k+1,col)}, k = (v<4 ? 2v : 16+2(v-4)) + 8*(lane>>4),
// col = ct*16 + (lane&15). One small block; all divergence lives here.
template <int F, int HFo>
__global__ void pack_weights(const float* __restrict__ W, const float* __restrict__ u,
                             unsigned int* __restrict__ pbuf) {
  constexpr int H    = HEADS;
  constexpr int Ctot = HFo + H;
  constexpr int CT   = (Ctot + 15) / 16;
  int t = threadIdx.x;                 // blockDim.x = 128 >= CT*32
  if (t >= CT * 32) return;
  int lane = t & 31;
  int ct   = t >> 5;
  int hi   = lane >> 4;
  int col  = ct * 16 + (lane & 15);
#pragma unroll
  for (int v = 0; v < 8; ++v) {
    int k = ((v < 4) ? 2 * v : 16 + 2 * (v - 4)) + 8 * hi;
    float b0 = 0.0f, b1 = 0.0f;
    if (col < Ctot) {
      if (k < F)
        b0 = (col < HFo) ? W[k * HFo + col] : u[k * H + (col - HFo)];
      if (k + 1 < F)
        b1 = (col < HFo) ? W[(k + 1) * HFo + col] : u[(k + 1) * H + (col - HFo)];
    }
    union { _Float16 h[2]; unsigned int d; } pk;
    pk.h[0] = (_Float16)b0;
    pk.h[1] = (_Float16)b1;
    pbuf[(ct * 32 + lane) * 8 + v] = pk.d;
  }
}

// ya[N x S] : cols [0,HFo) = x@W, cols [HFo,HFo+4) = x@u, cols >= HFo+4 zero.
// One wave = one 16-row node tile, K padded to 32, wave32 WMMA f32<-f16.
template <int F, int HFo>
__global__ __launch_bounds__(256)
void node_gemm_wmma(const float* __restrict__ x, const uint4* __restrict__ pbuf,
                    float* __restrict__ ya, int N) {
  constexpr int H    = HEADS;
  constexpr int Ctot = HFo + H;
  constexpr int CT   = (Ctot + 15) / 16;
  constexpr int S    = CT * 16;
  (void)H;

  int lane = threadIdx.x & 31;
  int wave = (blockIdx.x * blockDim.x + threadIdx.x) >> 5;
  int nTiles = (N + 15) >> 4;
  if (wave >= nTiles) return;          // wave-uniform: EXEC all-1 at WMMA
  int hi  = lane >> 4;                 // half-wave select
  int r   = lane & 15;
  int row = wave * 16 + r;
  int rclamp = row < N ? row : N - 1;  // unguarded (safe) A loads
  bool fullTile = (wave * 16 + 16) <= N;

  // A: 16x32 f16. Lane holds row (lane&15); VGPR v<4 -> K=2v(+1)+8*hi,
  // v>=4 -> K=16+2(v-4)(+1)+8*hi. K >= F folds to zero at compile time.
  v16h A;
#pragma unroll
  for (int v = 0; v < 8; ++v) {
    int k = ((v < 4) ? 2 * v : 16 + 2 * (v - 4)) + 8 * hi;
    float a0 = (k     < F) ? x[rclamp * F + k]     : 0.0f;
    float a1 = (k + 1 < F) ? x[rclamp * F + k + 1] : 0.0f;
    A[2 * v]     = (_Float16)a0;
    A[2 * v + 1] = (_Float16)a1;
  }

#pragma unroll
  for (int ct = 0; ct < CT; ++ct) {
    const int col = ct * 16 + r;
    // B: two coalesced 16B loads from the pre-packed lane-layout buffer.
    union { v16h h; uint4 q[2]; } B;
    const uint4* pb = pbuf + (size_t)(ct * 32 + lane) * 2;
    B.q[0] = pb[0];
    B.q[1] = pb[1];

    v8f Cz = {};
    v8f D = __builtin_amdgcn_wmma_f32_16x16x32_f16(
        /*neg_a=*/false, A, /*neg_b=*/false, B.h,
        /*c_mod=*/(short)0, Cz, /*reuse_a=*/false, /*reuse_b=*/false);
    // D layout: row = j + 8*hi (within tile), col = lane&15.
    // Unconditional coalesced stores (pad cols carry D==0).
    if (fullTile) {
#pragma unroll
      for (int j = 0; j < 8; ++j) {
        int rr = wave * 16 + j + 8 * hi;
        ya[(size_t)rr * S + col] = D[j];
      }
    } else {
#pragma unroll
      for (int j = 0; j < 8; ++j) {
        int rr = wave * 16 + j + 8 * hi;
        if (rr < N) ya[(size_t)rr * S + col] = D[j];
      }
    }
  }
}

// One thread per edge: q = softmax_h(a[src]-a[dst]+c); msg = q . y[src]; scatter to acc[dst].
template <int FO, int S>
__global__ __launch_bounds__(256)
void edge_msg(const int* __restrict__ ei, int E,
              const float* __restrict__ ya,
              const float* __restrict__ cvec, float* __restrict__ acc) {
  int e = blockIdx.x * blockDim.x + threadIdx.x;
  if (e >= E) return;
  int s = ei[e];
  int d = ei[E + e];
  const float* ys = ya + (size_t)s * S;       // y row of src
  float4 as = *reinterpret_cast<const float4*>(ys + 4 * FO);
  float4 ad = *reinterpret_cast<const float4*>(ya + (size_t)d * S + 4 * FO);
  float l0 = as.x - ad.x + cvec[0];
  float l1 = as.y - ad.y + cvec[1];
  float l2 = as.z - ad.z + cvec[2];
  float l3 = as.w - ad.w + cvec[3];
  float m  = fmaxf(fmaxf(l0, l1), fmaxf(l2, l3));
  float q0 = expf(l0 - m), q1 = expf(l1 - m), q2 = expf(l2 - m), q3 = expf(l3 - m);
  float inv = 1.0f / (q0 + q1 + q2 + q3);
  q0 *= inv; q1 *= inv; q2 *= inv; q3 *= inv;
#pragma unroll
  for (int o = 0; o < FO; ++o) {
    float v = q0 * ys[o] + q1 * ys[FO + o] + q2 * ys[2 * FO + o] + q3 * ys[3 * FO + o];
    atomicAdd(&acc[(size_t)d * FO + o], v);
  }
}

// Add self-loop message (q = softmax(c), xj = xi), divide by degree, + bias, ReLU.
template <int FO, int S>
__global__ __launch_bounds__(256)
void node_finalize(const float* __restrict__ acc, const float* __restrict__ ya,
                   const float* __restrict__ cvec, const float* __restrict__ bvec,
                   const float* __restrict__ cnt, float* __restrict__ h, int N) {
  int i = blockIdx.x * blockDim.x + threadIdx.x;
  if (i >= N) return;
  float c0 = cvec[0], c1 = cvec[1], c2 = cvec[2], c3 = cvec[3];
  float m  = fmaxf(fmaxf(c0, c1), fmaxf(c2, c3));
  float q0 = expf(c0 - m), q1 = expf(c1 - m), q2 = expf(c2 - m), q3 = expf(c3 - m);
  float inv = 1.0f / (q0 + q1 + q2 + q3);
  q0 *= inv; q1 *= inv; q2 *= inv; q3 *= inv;
  const float* yi = ya + (size_t)i * S;
  float invc = 1.0f / cnt[i];
#pragma unroll
  for (int o = 0; o < FO; ++o) {
    float v = acc[(size_t)i * FO + o]
            + q0 * yi[o] + q1 * yi[FO + o] + q2 * yi[2 * FO + o] + q3 * yi[3 * FO + o];
    v = v * invc + bvec[o];
    h[(size_t)i * FO + o] = fmaxf(v, 0.0f);
  }
}

// Per-channel sum / sumsq over nodes (8 channels). Registers -> LDS atomics -> global.
__global__ __launch_bounds__(256)
void bn_stats(const float* __restrict__ h, int N, float* __restrict__ stats) {
  __shared__ float ssum[8], ssq[8];
  if (threadIdx.x < 8) { ssum[threadIdx.x] = 0.0f; ssq[threadIdx.x] = 0.0f; }
  __syncthreads();
  float ls[8], lq[8];
#pragma unroll
  for (int o = 0; o < 8; ++o) { ls[o] = 0.0f; lq[o] = 0.0f; }
  for (int i = blockIdx.x * blockDim.x + threadIdx.x; i < N; i += gridDim.x * blockDim.x) {
#pragma unroll
    for (int o = 0; o < 8; ++o) {
      float v = h[(size_t)i * 8 + o];
      ls[o] += v; lq[o] += v * v;
    }
  }
#pragma unroll
  for (int o = 0; o < 8; ++o) { atomicAdd(&ssum[o], ls[o]); atomicAdd(&ssq[o], lq[o]); }
  __syncthreads();
  if (threadIdx.x < 8) {
    atomicAdd(&stats[threadIdx.x],     ssum[threadIdx.x]);
    atomicAdd(&stats[8 + threadIdx.x], ssq[threadIdx.x]);
  }
}

__global__ __launch_bounds__(256)
void bn_apply(const float* __restrict__ h, const float* __restrict__ stats,
              const float* __restrict__ gamma, const float* __restrict__ beta,
              float* __restrict__ out, int N) {
  int i = blockIdx.x * blockDim.x + threadIdx.x;
  if (i >= N) return;
  float invN = 1.0f / (float)N;
#pragma unroll
  for (int o = 0; o < 8; ++o) {
    float mean = stats[o] * invN;
    float var  = stats[8 + o] * invN - mean * mean;
    float v = (h[(size_t)i * 8 + o] - mean) * rsqrtf(var + 1e-5f) * gamma[o] + beta[o];
    out[(size_t)i * 8 + o] = v;
  }
}

extern "C" void kernel_launch(void* const* d_in, const int* in_sizes, int n_in,
                              void* d_out, int out_size, void* d_ws, size_t ws_size,
                              hipStream_t stream) {
  (void)n_in; (void)out_size; (void)ws_size;
  const float* x  = (const float*)d_in[0];
  const int*   ei = (const int*)d_in[1];
  const float* Wl[4] = {(const float*)d_in[2],  (const float*)d_in[6],
                        (const float*)d_in[10], (const float*)d_in[14]};
  const float* ul[4] = {(const float*)d_in[3],  (const float*)d_in[7],
                        (const float*)d_in[11], (const float*)d_in[15]};
  const float* cl[4] = {(const float*)d_in[4],  (const float*)d_in[8],
                        (const float*)d_in[12], (const float*)d_in[16]};
  const float* bl[4] = {(const float*)d_in[5],  (const float*)d_in[9],
                        (const float*)d_in[13], (const float*)d_in[17]};
  const float* gamma = (const float*)d_in[18];
  const float* beta  = (const float*)d_in[19];

  const int N = in_sizes[0] / 16;
  const int E = in_sizes[1] / 2;

  // Workspace carve-up (64B-aligned chunks)
  float* ws = (float*)d_ws;
  size_t off = 0;
  auto carve = [&](size_t nf) { float* p = ws + off; off += (nf + 15) & ~(size_t)15; return p; };
  float* cnt   = carve((size_t)N);
  float* yabuf = carve((size_t)N * 48);       // padded fused [y | a] rows (max stride 48)
  float* accb  = carve((size_t)N * 8);
  float* hA    = carve((size_t)N * 8);
  float* hB    = carve((size_t)N * 8);
  float* stats = carve(16);
  unsigned int* pbuf = (unsigned int*)carve(3 * 32 * 8);  // packed B operand (max CT=3)

  const int TB = 256;
  auto nb = [&](long long n) { return (int)((n + TB - 1) / TB); };
  const int nTiles = (N + 15) / 16;
  const int gemmBlocks = nb((long long)nTiles * 32);

  // Degrees (fresh every call; includes self-loop)
  fill_f32<<<nb(N), TB, 0, stream>>>(cnt, N, 1.0f);
  count_deg<<<nb(E), TB, 0, stream>>>(ei, E, cnt);

  // ---- Layer 1: F=16, Fo=4, S=32 (x -> hA)
  pack_weights<16, 16><<<1, 128, 0, stream>>>(Wl[0], ul[0], pbuf);
  node_gemm_wmma<16, 16><<<gemmBlocks, TB, 0, stream>>>(x, (const uint4*)pbuf, yabuf, N);
  fill_f32<<<nb((long long)N * 4), TB, 0, stream>>>(accb, N * 4, 0.0f);
  edge_msg<4, 32><<<nb(E), TB, 0, stream>>>(ei, E, yabuf, cl[0], accb);
  node_finalize<4, 32><<<nb(N), TB, 0, stream>>>(accb, yabuf, cl[0], bl[0], cnt, hA, N);

  // ---- Layer 2: F=4, Fo=4, S=32 (hA -> hB)
  pack_weights<4, 16><<<1, 128, 0, stream>>>(Wl[1], ul[1], pbuf);
  node_gemm_wmma<4, 16><<<gemmBlocks, TB, 0, stream>>>(hA, (const uint4*)pbuf, yabuf, N);
  fill_f32<<<nb((long long)N * 4), TB, 0, stream>>>(accb, N * 4, 0.0f);
  edge_msg<4, 32><<<nb(E), TB, 0, stream>>>(ei, E, yabuf, cl[1], accb);
  node_finalize<4, 32><<<nb(N), TB, 0, stream>>>(accb, yabuf, cl[1], bl[1], cnt, hB, N);

  // ---- Layer 3: F=4, Fo=4, S=32 (hB -> hA)
  pack_weights<4, 16><<<1, 128, 0, stream>>>(Wl[2], ul[2], pbuf);
  node_gemm_wmma<4, 16><<<gemmBlocks, TB, 0, stream>>>(hB, (const uint4*)pbuf, yabuf, N);
  fill_f32<<<nb((long long)N * 4), TB, 0, stream>>>(accb, N * 4, 0.0f);
  edge_msg<4, 32><<<nb(E), TB, 0, stream>>>(ei, E, yabuf, cl[2], accb);
  node_finalize<4, 32><<<nb(N), TB, 0, stream>>>(accb, yabuf, cl[2], bl[2], cnt, hA, N);

  // ---- Layer 4: F=4, Fo=8, S=48 (hA -> hB)
  pack_weights<4, 32><<<1, 128, 0, stream>>>(Wl[3], ul[3], pbuf);
  node_gemm_wmma<4, 32><<<gemmBlocks, TB, 0, stream>>>(hA, (const uint4*)pbuf, yabuf, N);
  fill_f32<<<nb((long long)N * 8), TB, 0, stream>>>(accb, N * 8, 0.0f);
  edge_msg<8, 48><<<nb(E), TB, 0, stream>>>(ei, E, yabuf, cl[3], accb);
  node_finalize<8, 48><<<nb(N), TB, 0, stream>>>(accb, yabuf, cl[3], bl[3], cnt, hB, N);

  // ---- BatchNorm (training-mode batch stats, biased var)
  fill_f32<<<1, 32, 0, stream>>>(stats, 16, 0.0f);
  bn_stats<<<256, TB, 0, stream>>>(hB, N, stats);
  bn_apply<<<nb(N), TB, 0, stream>>>(hB, stats, gamma, beta, (float*)d_out, N);
}
